// MultiAspectRetrieval_2190433321314
// MI455X (gfx1250) — compile-verified, matrix operand order checked by reading, not run on hardware
//
#include <hip/hip_runtime.h>

// ---------------------------------------------------------------------------
// MultiAspectRetrieval for MI455X (gfx1250, wave32, WMMA).
// S=8 aspects, d_k=64, d_A=2048, B=64, N=100000, K_MAX=32, is_warmup=0 path.
//
// s_i[b,n] = sum_s w_s * <qn[b,s,:], kn[s,n,:]>  done as bf16 WMMA GEMMs
// (w_s folded into A).  top-k(raw) == top-k(s_i) since raw is monotone in s_i.
// Key tiles are streamed HBM->LDS with the Tensor Data Mover, double-buffered
// so the DMA overlaps normalize+WMMA (kernel is HBM-bandwidth bound).
// ---------------------------------------------------------------------------

typedef __attribute__((ext_vector_type(16))) __bf16       v16bf;
typedef __attribute__((ext_vector_type(8)))  float        v8f;
typedef __attribute__((ext_vector_type(4)))  unsigned int v4u;
typedef __attribute__((ext_vector_type(8)))  int          v8i;
typedef __attribute__((ext_vector_type(4)))  int          v4i;

#define S_ASPECT 8
#define DK 64
#define DA 2048
#define BQ 64
#define KTOP 32
#define NEPS 1e-8f
#define NT 128          // keys per block tile

#if defined(__HIP_DEVICE_COMPILE__) && defined(__has_builtin)
#if __has_builtin(__builtin_amdgcn_tensor_load_to_lds) && __has_builtin(__builtin_amdgcn_s_wait_tensorcnt)
#define USE_TDM 1
#endif
#endif

__device__ __forceinline__ unsigned pack2bf(float a, float b) {
  union { __bf16 h[2]; unsigned u; } p;
  p.h[0] = (__bf16)a; p.h[1] = (__bf16)b;
  return p.u;
}

// ---------------------------------------------------------------------------
// Kernel A: queries = einsum('ska,ba->bsk'), L2-normalize over k, scale by
// softmax(aspect_weights)[s], emit bf16 in WMMA A-fragment layout:
//   frag element index = (((s*4 + mtile)*2 + khalf)*32 + lane)*16 + r*2 + e
// with A 16x32 bf16 layout: lane = half*16 + (b%16),
//   kk = khalf*32 + (r>=4?16:0) + half*8 + (r&3)*2 + e
// ---------------------------------------------------------------------------
__global__ void __launch_bounds__(64)
aspect_query_kernel(const float* __restrict__ z, const float* __restrict__ W_Q,
                    const float* __restrict__ aspect_w, __bf16* __restrict__ fragA) {
  const int k = threadIdx.x;          // 0..63
  const int b = blockIdx.x >> 3;
  const int s = blockIdx.x & 7;

  // softmax over 8 aspect weights (redundant per block; trivial)
  float m = -3.4e38f;
  for (int i = 0; i < S_ASPECT; i++) m = fmaxf(m, aspect_w[i]);
  float dsum = 0.f;
  for (int i = 0; i < S_ASPECT; i++) dsum += expf(aspect_w[i] - m);
  const float w_s = expf(aspect_w[s] - m) / dsum;

  const float4* wq4 = (const float4*)(W_Q + ((size_t)s * DK + k) * DA);
  const float4* z4  = (const float4*)(z + (size_t)b * DA);
  float acc = 0.f;
#pragma unroll 4
  for (int a = 0; a < DA / 4; a++) {
    float4 wv = wq4[a], zv = z4[a];
    acc += wv.x * zv.x + wv.y * zv.y + wv.z * zv.z + wv.w * zv.w;
  }

  __shared__ float q[DK];
  q[k] = acc;
  __syncthreads();
  float ss = 0.f;
  for (int j = 0; j < DK; j++) { float t = q[j]; ss += t * t; }
  const float val = acc * (w_s / (sqrtf(ss) + NEPS));

  const int mrow = b & 15, mtile = b >> 4;
  const int kh = k >> 5, wlo = k & 31;
  const int hi = (wlo >> 4) & 1, hp = (wlo >> 3) & 1;
  const int p = (wlo & 7) >> 1, e = wlo & 1;
  const int r = hi * 4 + p;
  const int lane = hp * 16 + mrow;
  const size_t idx = ((((size_t)s * 4 + mtile) * 2 + kh) * 32 + lane) * 16 + r * 2 + e;
  fragA[idx] = (__bf16)val;
}

// ---------------------------------------------------------------------------
// Kernel B: per 128-key tile, per aspect: TDM-prefetch raw f32 keys to LDS
// (double-buffered, overlapped with compute), normalize -> bf16 LDS tile
// (36-dword row stride), WMMA accumulate s_i.
// Block = 256 threads = 8 waves; wave w: M-tile = w&3, N-half = w>>2.
// ---------------------------------------------------------------------------
__global__ void __launch_bounds__(256)
similarity_kernel(const float* __restrict__ pool_keys, const __bf16* __restrict__ fragA,
                  float* __restrict__ s_i, int N) {
  __shared__ __align__(16) float    rawT[2][NT * 64];  // double-buffered raw f32 tiles
  __shared__ __align__(16) unsigned bT[NT * 36];       // bf16 tile, 36-dword row stride

  const int tid   = threadIdx.x;
  const int lane  = tid & 31;
  const int wv    = tid >> 5;      // wave id
  const int mtile = wv & 3;        // 16 batch rows
  const int nhalf = wv >> 2;       // which 64-key half of the tile
  const int nbase = blockIdx.x * NT;
  const int nloc  = lane & 15;
  const int lh    = lane >> 4;

  const v8f vzero = {0.f, 0.f, 0.f, 0.f, 0.f, 0.f, 0.f, 0.f};
  v8f acc[4];
#pragma unroll
  for (int t = 0; t < 4; t++) acc[t] = vzero;

#ifdef USE_TDM
  // ---- TDM tile loader: 2-D descriptor, OOB rows (> N) read back as zero ----
  auto issue_tile = [&](int s, int buf) {
    const unsigned ldsOff = (unsigned)(unsigned long long)(void*)&rawT[buf][0];
    unsigned long long ga =
        (unsigned long long)(const void*)(pool_keys + ((size_t)s * N + nbase) * DK);
    const unsigned remain = (unsigned)(N - nbase);     // rows available from tile start
    v4u g0;
    g0.x = 1u;                                         // count = 1 valid descriptor
    g0.y = ldsOff;                                     // lds_addr
    g0.z = (unsigned)ga;                               // global_addr[31:0]
    g0.w = ((unsigned)(ga >> 32) & 0x1FFFFFFu) | (2u << 30);   // addr hi | type=2
    v8i g1;
    g1[0] = 0x20000;                                   // data_size = 2 (4 bytes)
    g1[1] = (int)(64u << 16);                          // tensor_dim0 = 64
    g1[2] = (int)((remain & 0xFFFFu) << 16);           // td0 hi=0 | tensor_dim1 lo
    g1[3] = (int)((remain >> 16) | (64u << 16));       // tensor_dim1 hi | tile_dim0 = 64
    g1[4] = NT;                                        // tile_dim1 = 128, tile_dim2 = 0
    g1[5] = 64;                                        // tensor_dim0_stride = 64
    g1[6] = 0;
    g1[7] = 0;
    v4i g2 = {0, 0, 0, 0}, g3 = {0, 0, 0, 0};
    v8i g4 = {0, 0, 0, 0, 0, 0, 0, 0};                 // unused trailing group (2-D)
    __builtin_amdgcn_tensor_load_to_lds(g0, g1, g2, g3, g4, 0);
  };
  if (wv == 0) issue_tile(0, 0);                       // prefetch aspect 0
#endif

  for (int s = 0; s < S_ASPECT; s++) {
#ifdef USE_TDM
    if (wv == 0) {
      if (s + 1 < S_ASPECT) {
        issue_tile(s + 1, (s + 1) & 1);                // prefetch next aspect
        __builtin_amdgcn_s_wait_tensorcnt(1);          // in-order: tile s has landed
      } else {
        __builtin_amdgcn_s_wait_tensorcnt(0);
      }
    }
#else
    {
      const int j2 = tid >> 1, h2 = tid & 1;
      const float4* src =
          (const float4*)(pool_keys + ((size_t)s * N + nbase + j2) * DK + h2 * 32);
      float4* dst = (float4*)&rawT[s & 1][j2 * 64 + h2 * 32];
      const bool ok = (nbase + j2) < N;
#pragma unroll
      for (int i = 0; i < 8; i++) dst[i] = ok ? src[i] : make_float4(0.f, 0.f, 0.f, 0.f);
    }
#endif
    __syncthreads();

    // ---- normalize rows, convert to bf16 tile (2 threads per key) ----
    {
      const int j2 = tid >> 1, h2 = tid & 1;
      const float4* src = (const float4*)&rawT[s & 1][j2 * 64 + h2 * 32];
      float4 vbuf[8];
      float ssq = 0.f;
#pragma unroll
      for (int i = 0; i < 8; i++) {
        float4 v = src[i]; vbuf[i] = v;
        ssq += v.x * v.x + v.y * v.y + v.z * v.z + v.w * v.w;
      }
      ssq += __shfl_xor(ssq, 1);              // partner thread (same key)
      const float scale = 1.0f / (sqrtf(ssq) + NEPS);
      uint2* dst = (uint2*)bT;
#pragma unroll
      for (int i = 0; i < 8; i++) {
        float4 v = vbuf[i];
        dst[j2 * 18 + h2 * 8 + i] =
            make_uint2(pack2bf(v.x * scale, v.y * scale),
                       pack2bf(v.z * scale, v.w * scale));
      }
    }
    __syncthreads();

    // ---- WMMA: A = w_s*qn[s] (16 batch rows), B = kn tile columns ----
    const v16bf* ap = (const v16bf*)fragA;
    const v16bf a0 = ap[(((size_t)s * 4 + mtile) * 2 + 0) * 32 + lane];
    const v16bf a1 = ap[(((size_t)s * 4 + mtile) * 2 + 1) * 32 + lane];

    const uint4* bq = (const uint4*)bT;
#pragma unroll
    for (int t = 0; t < 4; t++) {
      const int keyrow = nhalf * 64 + t * 16 + nloc;
      const int base = keyrow * 36 + lh * 4;          // dword index, 16B aligned
      union { v16bf v; uint4 q[2]; } bf0, bf1;
      bf0.q[0] = bq[(base) >> 2];
      bf0.q[1] = bq[(base + 8) >> 2];
      bf1.q[0] = bq[(base + 16) >> 2];
      bf1.q[1] = bq[(base + 24) >> 2];
      acc[t] = __builtin_amdgcn_wmma_f32_16x16x32_bf16(
          false, a0, false, bf0.v, (short)0, acc[t], false, false);
      acc[t] = __builtin_amdgcn_wmma_f32_16x16x32_bf16(
          false, a1, false, bf1.v, (short)0, acc[t], false, false);
    }
    __syncthreads();
  }

  // ---- store s_i: C layout -> row = 16*mtile + 8*lh + r ----
#pragma unroll
  for (int t = 0; t < 4; t++) {
    const int n = nbase + nhalf * 64 + t * 16 + nloc;
    if (n < N) {
#pragma unroll
      for (int r = 0; r < 8; r++) {
        const int brow = mtile * 16 + lh * 8 + r;
        s_i[(size_t)brow * N + n] = acc[t][r];
      }
    }
  }
}

// ---------------------------------------------------------------------------
// Kernel C: per batch row, top-32 of s_i (monotone transform => same set as
// top-32 of raw), exact reference normalization incl. EPS*(D+EPS) term.
// ---------------------------------------------------------------------------
__global__ void __launch_bounds__(256)
topk_kernel(const float* __restrict__ s_i, const float* __restrict__ tau_p,
            const float* __restrict__ lam_p, float* __restrict__ alpha_out,
            int* __restrict__ idx_out, int N) {
  const int b = blockIdx.x, tid = threadIdx.x;
  const float tau = tau_p[0], lam = lam_p[0];
  const float* row = s_i + (size_t)b * N;

  float lv[KTOP]; int li[KTOP];
#pragma unroll
  for (int j = 0; j < KTOP; j++) { lv[j] = -3.4e38f; li[j] = 0; }
  float rowsum = 0.f;

  for (int n = tid; n < N; n += 256) {
    const float v = row[n];
    const float g = 1.f / (1.f + expf(-lam * (v - tau)));
    rowsum += g * expf(v);                       // TEMP == 1
    if (v > lv[KTOP - 1]) {
      int pos = 0;
#pragma unroll
      for (int j = 0; j < KTOP; j++) pos += (lv[j] >= v) ? 1 : 0;
#pragma unroll
      for (int j = KTOP - 1; j > 0; j--)
        if (j > pos) { lv[j] = lv[j - 1]; li[j] = li[j - 1]; }
#pragma unroll
      for (int j = 0; j < KTOP; j++)
        if (j == pos) { lv[j] = v; li[j] = n; }
    }
  }

  __shared__ float sv[256 * KTOP];
  __shared__ int   sx[256 * KTOP];
  __shared__ float red[256];
#pragma unroll
  for (int j = 0; j < KTOP; j++) { sv[tid * KTOP + j] = lv[j]; sx[tid * KTOP + j] = li[j]; }
  red[tid] = rowsum;
  __syncthreads();

  for (int halfc = 128; halfc >= 1; halfc >>= 1) {
    if (tid < halfc) {
      const int a0 = tid * KTOP, b0 = (tid + halfc) * KTOP;
      int pa = 0, pb = 0;
      float mv[KTOP]; int mi[KTOP];
#pragma unroll
      for (int j = 0; j < KTOP; j++) {
        const float av = sv[a0 + (pa & (KTOP - 1))];
        const int   ai = sx[a0 + (pa & (KTOP - 1))];
        const float bv = sv[b0 + (pb & (KTOP - 1))];
        const int   bi = sx[b0 + (pb & (KTOP - 1))];
        bool ta;
        if (pa >= KTOP) ta = false;
        else if (pb >= KTOP) ta = true;
        else ta = (av > bv) || (av == bv && ai <= bi);
        mv[j] = ta ? av : bv; mi[j] = ta ? ai : bi;
        pa += ta ? 1 : 0; pb += ta ? 0 : 1;
      }
#pragma unroll
      for (int j = 0; j < KTOP; j++) { sv[a0 + j] = mv[j]; sx[a0 + j] = mi[j]; }
    }
    __syncthreads();
  }

  if (tid == 0) {
    float t = 0.f;
    for (int i = 0; i < 256; i++) t += red[i];
    red[0] = t;
  }
  __syncthreads();
  const float D = red[0];

  if (tid < KTOP) {
    const float v = sv[tid];
    const float g = 1.f / (1.f + expf(-lam * (v - tau)));
    red[64 + tid] = g * expf(v);
  }
  __syncthreads();
  if (tid < KTOP) {
    float tsum = 0.f;
#pragma unroll
    for (int j = 0; j < KTOP; j++) tsum += red[64 + j];
    const float v = sv[tid];
    const float g = 1.f / (1.f + expf(-lam * (v - tau)));
    const float raw = g * expf(v);
    alpha_out[b * KTOP + tid] = raw / (tsum + NEPS * (D + NEPS));
    idx_out[b * KTOP + tid] = sx[tid];
  }
}

// ---------------------------------------------------------------------------
extern "C" void kernel_launch(void* const* d_in, const int* in_sizes, int n_in,
                              void* d_out, int out_size, void* d_ws, size_t ws_size,
                              hipStream_t stream) {
  const float* z          = (const float*)d_in[0];
  const float* pool_keys  = (const float*)d_in[1];
  const float* W_Q        = (const float*)d_in[2];
  const float* aspect_w   = (const float*)d_in[3];
  const float* tau        = (const float*)d_in[4];
  const float* lam        = (const float*)d_in[5];
  // d_in[6] = is_warmup, 0 in this configuration -> gated/renormalized path.

  const int N = in_sizes[1] / (S_ASPECT * DK);   // 100000

  char* ws = (char*)d_ws;
  __bf16* fragA = (__bf16*)ws;                   // 8*4*2*32*16 bf16 = 64 KiB
  float* s_i = (float*)(ws + 65536);             // 64 * N f32 (~25.6 MB)

  aspect_query_kernel<<<dim3(BQ * S_ASPECT), dim3(64), 0, stream>>>(z, W_Q, aspect_w, fragA);
  similarity_kernel<<<dim3((N + NT - 1) / NT), dim3(256), 0, stream>>>(pool_keys, fragA, s_i, N);
  topk_kernel<<<dim3(BQ), dim3(256), 0, stream>>>(s_i, tau, lam, (float*)d_out,
                                                  (int*)d_out + BQ * KTOP, N);
}
